// SA_Lite_14465449853071
// MI455X (gfx1250) — matrix operand order repaired and checked by reading, not compiled, not run on hardware
//
#include <hip/hip_runtime.h>
#include <hip/hip_bf16.h>

typedef __attribute__((ext_vector_type(16))) _Float16 v16h;
typedef __attribute__((ext_vector_type(8)))  _Float16 v8h;
typedef __attribute__((ext_vector_type(8)))  float    v8f;

#define BATCH  16
#define NPTS   4096
#define NPOINT 1024
#define KNN    32
#define CIN    67
#define CPAD   96   // 67 padded to 3 K-chunks of 32
#define C1     64
#define C3     128

// ---------------------------------------------------------------------------
// Kernel 1: farthest point sampling, one workgroup per batch.
// Writes cent_idx[b][it] and new_xyz (first part of d_out).
// ---------------------------------------------------------------------------
__global__ __launch_bounds__(256) void fps_kernel(const float* __restrict__ xyz,
                                                  int* __restrict__ cent,
                                                  float* __restrict__ new_xyz) {
    __shared__ float dmin[NPTS];
    __shared__ float rv[256];
    __shared__ int   ri[256];

    const int b   = blockIdx.x;
    const int tid = threadIdx.x;
    const float* xb = xyz + (size_t)b * NPTS * 3;

    for (int i = tid; i < NPTS; i += 256) dmin[i] = 1e10f;
    __syncthreads();

    int far = 0;
    for (int it = 0; it < NPOINT; ++it) {
        if (tid < 3) new_xyz[((size_t)b * NPOINT + it) * 3 + tid] = xb[far * 3 + tid];
        if (tid == 0) cent[b * NPOINT + it] = far;

        const float cx = xb[far * 3 + 0];
        const float cy = xb[far * 3 + 1];
        const float cz = xb[far * 3 + 2];

        float best = -1.0f;
        int   bi   = 0;
        for (int i = tid; i < NPTS; i += 256) {
            float dx = xb[i * 3 + 0] - cx;
            float dy = xb[i * 3 + 1] - cy;
            float dz = xb[i * 3 + 2] - cz;
            float d  = dx * dx + dy * dy + dz * dz;
            float nd = fminf(dmin[i], d);
            dmin[i]  = nd;
            if (nd > best) { best = nd; bi = i; }   // first-max tie-break (i ascending)
        }
        rv[tid] = best; ri[tid] = bi;
        __syncthreads();
        for (int off = 128; off > 0; off >>= 1) {
            if (tid < off) {
                float ov = rv[tid + off]; int oi = ri[tid + off];
                if (ov > rv[tid] || (ov == rv[tid] && oi < ri[tid])) { rv[tid] = ov; ri[tid] = oi; }
            }
            __syncthreads();
        }
        far = ri[0];
        __syncthreads();
    }
}

// ---------------------------------------------------------------------------
// Kernel 2: 32 nearest neighbors per centroid, one workgroup per centroid.
// ---------------------------------------------------------------------------
__global__ __launch_bounds__(128) void knn_kernel(const float* __restrict__ xyz,
                                                  const int* __restrict__ cent,
                                                  int* __restrict__ knn) {
    __shared__ float d2s[NPTS];
    __shared__ float rv[128];
    __shared__ int   ri[128];

    const int g   = blockIdx.x;           // 0 .. B*NPOINT-1
    const int b   = g / NPOINT;
    const int tid = threadIdx.x;
    const float* xb = xyz + (size_t)b * NPTS * 3;

    const int ci = cent[g];
    const float cx = xb[ci * 3 + 0];
    const float cy = xb[ci * 3 + 1];
    const float cz = xb[ci * 3 + 2];

    for (int i = tid; i < NPTS; i += 128) {
        float dx = xb[i * 3 + 0] - cx;
        float dy = xb[i * 3 + 1] - cy;
        float dz = xb[i * 3 + 2] - cz;
        d2s[i] = dx * dx + dy * dy + dz * dz;
    }
    __syncthreads();

    for (int r = 0; r < KNN; ++r) {
        float best = 3.4e38f;
        int   bi   = 0;
        for (int i = tid; i < NPTS; i += 128) {
            float v = d2s[i];
            if (v < best) { best = v; bi = i; }
        }
        rv[tid] = best; ri[tid] = bi;
        __syncthreads();
        for (int off = 64; off > 0; off >>= 1) {
            if (tid < off) {
                float ov = rv[tid + off]; int oi = ri[tid + off];
                if (ov < rv[tid] || (ov == rv[tid] && oi < ri[tid])) { rv[tid] = ov; ri[tid] = oi; }
            }
            __syncthreads();
        }
        if (tid == 0) {
            knn[(size_t)g * KNN + r] = ri[0];
            d2s[ri[0]] = 3.4e38f;
        }
        __syncthreads();
    }
}

// ---------------------------------------------------------------------------
// Kernel 3: fused gather + 3-layer MLP (WMMA f16->f32) + max over K, one wave
// per centroid, 2 waves (64 threads) per workgroup.  Static LDS = 56 KB.
// ---------------------------------------------------------------------------
__global__ __launch_bounds__(64) void mlp_kernel(const float* __restrict__ xyz,
                                                 const float* __restrict__ points,
                                                 const float* __restrict__ W1, const float* __restrict__ B1,
                                                 const float* __restrict__ W2, const float* __restrict__ B2,
                                                 const float* __restrict__ W3, const float* __restrict__ B3,
                                                 const int* __restrict__ knn,
                                                 float* __restrict__ out_points) {
    // Weights transposed to [out][in] so B-fragments are 16 contiguous halves.
    __shared__ __align__(32) _Float16 w1t[C1][CPAD];        // 12 KB (K padded, zero-filled)
    __shared__ __align__(32) _Float16 w2t[C1][C1];          //  8 KB
    __shared__ __align__(32) _Float16 w3t[C3][C1];          // 16 KB
    __shared__ __align__(32) _Float16 a0[2][KNN][CPAD];     // 12 KB, reused as h2 (stride CPAD)
    __shared__ __align__(32) _Float16 h1[2][KNN][C1];       //  8 KB

    const int tid = threadIdx.x;

    // ---- stage weights as f16 ----
    for (int i = tid; i < C1 * CPAD; i += 64) {
        int o = i / CPAD, c = i % CPAD;
        w1t[o][c] = (c < CIN) ? (_Float16)W1[o * CIN + c] : (_Float16)0.0f;
    }
    for (int i = tid; i < C1 * C1; i += 64)  w2t[i / C1][i % C1] = (_Float16)W2[i];
    for (int i = tid; i < C3 * C1; i += 64)  w3t[i / C1][i % C1] = (_Float16)W3[i];
    __syncthreads();

    const int wv   = tid >> 5;            // wave in block
    const int lane = tid & 31;
    const int lo   = lane & 15;
    const int hi   = lane >> 4;
    const int g    = blockIdx.x * 2 + wv; // centroid id
    const int b    = g / NPOINT;
    const int s    = g % NPOINT;

    // ---- gather grouped inputs: lane = neighbor row ----
    {
        const int p = knn[(size_t)g * KNN + lane];
        const float* xp = xyz + ((size_t)b * NPTS + p) * 3;
        a0[wv][lane][0] = (_Float16)xp[0];
        a0[wv][lane][1] = (_Float16)xp[1];
        a0[wv][lane][2] = (_Float16)xp[2];
        const float* pp = points + (size_t)b * C1 * NPTS + p;
        #pragma unroll 4
        for (int c = 0; c < C1; ++c) a0[wv][lane][3 + c] = (_Float16)pp[(size_t)c * NPTS];
        for (int c = CIN; c < CPAD; ++c) a0[wv][lane][c] = (_Float16)0.0f;
    }
    __syncthreads();

    // Fragment loaders: contiguous k order (k = 16*hi + s), same for A and B.
    auto loadA = [&](const _Float16* base, int stride, int m, int kk) -> v16h {
        return *(const v16h*)(base + (size_t)(m * 16 + lo) * stride + kk * 32 + hi * 16);
    };
    auto loadB = [&](const _Float16* base, int stride, int nt, int kk) -> v16h {
        return *(const v16h*)(base + (size_t)(nt * 16 + lo) * stride + kk * 32 + hi * 16);
    };

    // ---- layer 1: (32 x 96) x (96 x 64), bias in C, ReLU ----
    #pragma unroll
    for (int m = 0; m < 2; ++m) {
        #pragma unroll
        for (int nt = 0; nt < 4; ++nt) {
            const float bv = B1[nt * 16 + lo];
            v8f acc;
            #pragma unroll
            for (int r = 0; r < 8; ++r) acc[r] = bv;
            #pragma unroll
            for (int kk = 0; kk < 3; ++kk) {
                v16h a = loadA(&a0[wv][0][0], CPAD, m, kk);
                v16h w = loadB(&w1t[0][0],    CPAD, nt, kk);
                acc = __builtin_amdgcn_wmma_f32_16x16x32_f16(false, a, false, w,
                                                             (short)0, acc, false, false);
            }
            #pragma unroll
            for (int r = 0; r < 8; ++r) {
                float v = acc[r]; v = v > 0.0f ? v : 0.0f;
                h1[wv][m * 16 + hi * 8 + r][nt * 16 + lo] = (_Float16)v;
            }
        }
    }
    __syncthreads();

    // ---- layer 2: (32 x 64) x (64 x 64) -> h2 stored into a0 region ----
    #pragma unroll
    for (int m = 0; m < 2; ++m) {
        #pragma unroll
        for (int nt = 0; nt < 4; ++nt) {
            const float bv = B2[nt * 16 + lo];
            v8f acc;
            #pragma unroll
            for (int r = 0; r < 8; ++r) acc[r] = bv;
            #pragma unroll
            for (int kk = 0; kk < 2; ++kk) {
                v16h a = loadA(&h1[wv][0][0], C1, m, kk);
                v16h w = loadB(&w2t[0][0],    C1, nt, kk);
                acc = __builtin_amdgcn_wmma_f32_16x16x32_f16(false, a, false, w,
                                                             (short)0, acc, false, false);
            }
            #pragma unroll
            for (int r = 0; r < 8; ++r) {
                float v = acc[r]; v = v > 0.0f ? v : 0.0f;
                a0[wv][m * 16 + hi * 8 + r][nt * 16 + lo] = (_Float16)v;  // h2
            }
        }
    }
    __syncthreads();

    // ---- layer 3: (32 x 64) x (64 x 128), ReLU, max over the 32 rows ----
    #pragma unroll
    for (int nt = 0; nt < 8; ++nt) {
        const float bv = B3[nt * 16 + lo];
        float vmax = 0.0f;                       // ReLU outputs are >= 0
        #pragma unroll
        for (int m = 0; m < 2; ++m) {
            v8f acc;
            #pragma unroll
            for (int r = 0; r < 8; ++r) acc[r] = bv;
            #pragma unroll
            for (int kk = 0; kk < 2; ++kk) {
                v16h a = loadA(&a0[wv][0][0], CPAD, m, kk);   // h2, stride CPAD
                v16h w = loadB(&w3t[0][0],    C1,   nt, kk);
                acc = __builtin_amdgcn_wmma_f32_16x16x32_f16(false, a, false, w,
                                                             (short)0, acc, false, false);
            }
            #pragma unroll
            for (int r = 0; r < 8; ++r) {
                float v = acc[r]; v = v > 0.0f ? v : 0.0f;
                vmax = fmaxf(vmax, v);
            }
        }
        // lanes L and L+16 hold the same output channel for complementary rows
        float other = __shfl_xor(vmax, 16, 32);
        vmax = fmaxf(vmax, other);
        if (lane < 16) {
            const int ch = nt * 16 + lane;
            out_points[((size_t)b * C3 + ch) * NPOINT + s] = vmax;
        }
    }
}

// ---------------------------------------------------------------------------
extern "C" void kernel_launch(void* const* d_in, const int* in_sizes, int n_in,
                              void* d_out, int out_size, void* d_ws, size_t ws_size,
                              hipStream_t stream) {
    (void)in_sizes; (void)n_in; (void)out_size; (void)ws_size;
    const float* xyz    = (const float*)d_in[0];
    const float* points = (const float*)d_in[1];
    const float* W1     = (const float*)d_in[2];
    const float* B1     = (const float*)d_in[3];
    const float* W2     = (const float*)d_in[4];
    const float* B2     = (const float*)d_in[5];
    const float* W3     = (const float*)d_in[6];
    const float* B3     = (const float*)d_in[7];

    float* out        = (float*)d_out;
    float* new_xyz    = out;                              // B*NPOINT*3
    float* new_points = out + (size_t)BATCH * NPOINT * 3; // B*C3*NPOINT

    int* cent = (int*)d_ws;                 // B*NPOINT ints
    int* knn  = cent + BATCH * NPOINT;      // B*NPOINT*KNN ints

    fps_kernel<<<BATCH, 256, 0, stream>>>(xyz, cent, new_xyz);
    knn_kernel<<<BATCH * NPOINT, 128, 0, stream>>>(xyz, cent, knn);
    mlp_kernel<<<BATCH * NPOINT / 2, 64, 0, stream>>>(xyz, points,
                                                      W1, B1, W2, B2, W3, B3,
                                                      knn, new_points);
}
